// SimVQ_48919677501489
// MI455X (gfx1250) — matrix operand order repaired and checked by reading, not compiled, not run on hardware
//
#include <hip/hip_runtime.h>
#include <hip/hip_bf16.h>

typedef __bf16 bf16_t;
typedef __bf16 v8bf  __attribute__((ext_vector_type(8)));
typedef __bf16 v16bf __attribute__((ext_vector_type(16)));
typedef float  v8f   __attribute__((ext_vector_type(8)));

#define Bsz 8
#define Tsz 2048
#define Dd  128
#define Cc  8192
#define NROWS (Bsz * Tsz)                      /* 16384 */
#define COMMIT_SCALE (1.25f / (float)(NROWS * Dd))
#define NT 4                                   /* 16-row M tiles per wave (64 rows) */

// ---------------------------------------------------------------------------
// Kernel 1: weight-norm conv weight rows  w[d,:] = g[d] * v[d,:] / ||v[d,:]||
// ---------------------------------------------------------------------------
__global__ void k_build_w(const float* __restrict__ v, const float* __restrict__ g,
                          float* __restrict__ w) {
  int d = threadIdx.x;
  if (d >= Dd) return;
  const float* vr = v + (size_t)d * Dd;
  float s = 0.f;
  for (int k = 0; k < Dd; ++k) s += vr[k] * vr[k];
  float inv = g[d] / sqrtf(s);
  float* wr = w + (size_t)d * Dd;
  for (int k = 0; k < Dd; ++k) wr[k] = vr[k] * inv;
}

// ---------------------------------------------------------------------------
// Kernel 2: zh = bf16(-2 * z)   (fold the -2 of the distance into the A matrix)
// ---------------------------------------------------------------------------
__global__ void k_zconv(const float* __restrict__ z, bf16_t* __restrict__ zh) {
  int i = blockIdx.x * blockDim.x + threadIdx.x;
  zh[i] = (bf16_t)(-2.0f * z[i]);
}

// ---------------------------------------------------------------------------
// Kernel 3: implicit codebook cb = fc @ w.T + b  (f32 + bf16 copy)
// ---------------------------------------------------------------------------
__global__ void k_build_cb(const float* __restrict__ fc, const float* __restrict__ w,
                           const float* __restrict__ b,
                           float* __restrict__ cb, bf16_t* __restrict__ cbh) {
  int tid = blockIdx.x * blockDim.x + threadIdx.x;
  int c = tid >> 7, d = tid & 127;
  const float4* fr = (const float4*)(fc + (size_t)c * Dd);
  const float4* wr = (const float4*)(w + (size_t)d * Dd);
  float acc = 0.f;
#pragma unroll
  for (int k = 0; k < Dd / 4; ++k) {
    float4 a = fr[k], x = wr[k];
    acc += a.x * x.x + a.y * x.y + a.z * x.z + a.w * x.w;
  }
  acc += b[d];
  cb[tid]  = acc;
  cbh[tid] = (bf16_t)acc;
}

// ---------------------------------------------------------------------------
// Kernel 4: cbn2[c] = ||cb[c,:]||^2   (one wave32 per row)
// ---------------------------------------------------------------------------
__global__ void k_cbn2(const float* __restrict__ cb, float* __restrict__ cbn2) {
  int wave = (blockIdx.x * blockDim.x + threadIdx.x) >> 5;
  int lane = threadIdx.x & 31;
  const float* r = cb + (size_t)wave * Dd;
  float s = 0.f;
#pragma unroll
  for (int j = 0; j < 4; ++j) { float x = r[lane + j * 32]; s += x * x; }
  for (int off = 16; off >= 1; off >>= 1) s += __shfl_xor(s, off, 32);
  if (lane == 0) cbn2[wave] = s;
}

// ---------------------------------------------------------------------------
// Kernel 5: WMMA distance + argmin.
//   One wave handles 64 z-rows (4 M tiles), scans 512 code tiles with a
//   register-double-buffered B stream (prefetch next tile during WMMA).
//   score[m][n] = cbn2[n] + sum_k (-2 z[m,k]) cb[n,k]   (C seeded with cbn2)
// ---------------------------------------------------------------------------
__device__ __forceinline__ void load_btile(const bf16_t* __restrict__ cbh,
                                           const float* __restrict__ cbn2,
                                           int tile, int nl, int half,
                                           v16bf Bm[4], float& cn) {
  const int n = tile * 16 + nl;
  const bf16_t* cr = cbh + (size_t)n * Dd;
#pragma unroll
  for (int kb = 0; kb < 4; ++kb) {
    v8bf lo = *(const v8bf*)(cr + kb * 32 + half * 8);
    v8bf hi = *(const v8bf*)(cr + kb * 32 + 16 + half * 8);
    Bm[kb] = __builtin_shufflevector(lo, hi, 0, 1, 2, 3, 4, 5, 6, 7,
                                     8, 9, 10, 11, 12, 13, 14, 15);
  }
  cn = cbn2[n];
}

__device__ __forceinline__ void compute_tile(const v16bf A[NT][4], const v16bf Bm[4],
                                             float cn, int n,
                                             float best[NT][8], int bidx[NT][8]) {
  v8f acc[NT];
#pragma unroll
  for (int t = 0; t < NT; ++t) acc[t] = (v8f){cn, cn, cn, cn, cn, cn, cn, cn};
  // kb-outer / t-inner: consecutive WMMAs are independent (no RAW chain)
#pragma unroll
  for (int kb = 0; kb < 4; ++kb)
#pragma unroll
    for (int t = 0; t < NT; ++t)
      acc[t] = __builtin_amdgcn_wmma_f32_16x16x32_bf16(
          false, A[t][kb], false, Bm[kb], (short)0, acc[t], false, false);
#pragma unroll
  for (int t = 0; t < NT; ++t)
#pragma unroll
    for (int r = 0; r < 8; ++r) {
      float s = acc[t][r];
      if (s < best[t][r]) { best[t][r] = s; bidx[t][r] = n; }  // strict < keeps lowest idx
    }
}

__global__ __launch_bounds__(32)
void k_argmin(const bf16_t* __restrict__ zh, const bf16_t* __restrict__ cbh,
              const float* __restrict__ cbn2,
              int* __restrict__ idx_out, float* __restrict__ fidx_out) {
  const int lane = threadIdx.x;
  const int half = lane >> 4;       // K-half select per WMMA layout
  const int nl   = lane & 15;       // row (A) / column (B,C,D) within tile
  const int rowBase = blockIdx.x * (16 * NT);

  // --- A operands: 16-bit A 16x32 layout; lanes 0-15: K={0..7,16..23}, 16-31: K={8..15,24..31}
  v16bf A[NT][4];
#pragma unroll
  for (int t = 0; t < NT; ++t) {
    const bf16_t* zr = zh + (size_t)(rowBase + t * 16 + nl) * Dd;
#pragma unroll
    for (int kb = 0; kb < 4; ++kb) {
      v8bf lo = *(const v8bf*)(zr + kb * 32 + half * 8);
      v8bf hi = *(const v8bf*)(zr + kb * 32 + 16 + half * 8);
      A[t][kb] = __builtin_shufflevector(lo, hi, 0, 1, 2, 3, 4, 5, 6, 7,
                                         8, 9, 10, 11, 12, 13, 14, 15);
    }
  }

  float best[NT][8];
  int   bidx[NT][8];
#pragma unroll
  for (int t = 0; t < NT; ++t)
#pragma unroll
    for (int r = 0; r < 8; ++r) { best[t][r] = 3.402823e38f; bidx[t][r] = 0; }

  // --- double-buffered B stream over 512 code tiles
  v16bf B0[4], B1[4];
  float cn0, cn1;
  load_btile(cbh, cbn2, 0, nl, half, B0, cn0);
  for (int tile = 0; tile < Cc / 16; tile += 2) {
    load_btile(cbh, cbn2, tile + 1, nl, half, B1, cn1);       // prefetch odd tile
    compute_tile(A, B0, cn0, tile * 16 + nl, best, bidx);
    load_btile(cbh, cbn2, tile + 2, nl, half, B0, cn0);       // prefetch next even (padded at 512)
    compute_tile(A, B1, cn1, (tile + 1) * 16 + nl, best, bidx);
  }

  // --- cross-lane argmin: D layout lane=col n, VGPR r = row r + 8*half
  __shared__ float ls[NT][16][16];
  __shared__ int   li[NT][16][16];
#pragma unroll
  for (int t = 0; t < NT; ++t)
#pragma unroll
    for (int r = 0; r < 8; ++r) {
      int m = r + 8 * half;
      ls[t][m][nl] = best[t][r];
      li[t][m][nl] = bidx[t][r];
    }
  __syncthreads();
#pragma unroll
  for (int tt = 0; tt < NT / 2; ++tt) {
    int t = tt * 2 + half, m = nl;  // 32 lanes x 2 passes cover the 64 rows
    float bs = ls[t][m][0];
    int   bi = li[t][m][0];
#pragma unroll
    for (int j = 1; j < 16; ++j) {
      float s = ls[t][m][j]; int i2 = li[t][m][j];
      if (s < bs || (s == bs && i2 < bi)) { bs = s; bi = i2; }
    }
    int row = rowBase + t * 16 + m;
    idx_out[row]  = bi;
    fidx_out[row] = (float)bi;
  }
}

// ---------------------------------------------------------------------------
// Kernel 6: gather z_q, rotation trick, per-row commit partial (one wave/row)
// ---------------------------------------------------------------------------
__global__ void k_finalize(const float* __restrict__ z, const float* __restrict__ cb,
                           const int* __restrict__ idx,
                           float* __restrict__ zq_out, float* __restrict__ commit_ws) {
  int row  = blockIdx.x * (blockDim.x >> 5) + (threadIdx.x >> 5);
  int lane = threadIdx.x & 31;
  const float4 zv = ((const float4*)(z + (size_t)row * Dd))[lane];
  int ci = idx[row];
  const float4 qv = ((const float4*)(cb + (size_t)ci * Dd))[lane];

  float szz = zv.x * zv.x + zv.y * zv.y + zv.z * zv.z + zv.w * zv.w;
  float sqq = qv.x * qv.x + qv.y * qv.y + qv.z * qv.z + qv.w * qv.w;
  float szq = zv.x * qv.x + zv.y * qv.y + zv.z * qv.z + zv.w * qv.w;
  for (int off = 16; off >= 1; off >>= 1) {
    szz += __shfl_xor(szz, off, 32);
    sqq += __shfl_xor(sqq, off, 32);
    szq += __shfl_xor(szq, off, 32);
  }
  const float EPSv = 1e-12f;
  float nsrc = sqrtf(szz), ntgt = sqrtf(sqq);
  float a  = fmaxf(nsrc, EPSv), bb = fmaxf(ntgt, EPSv);
  // u = z/a, q = zq/bb, w = (u+q)/wn
  float wn  = fmaxf(sqrtf(szz / (a * a) + 2.f * szq / (a * bb) + sqq / (bb * bb)), EPSv);
  float dew = (szz / a + szq / bb) / wn;   // e.w
  float deu = szz / a;                     // e.u
  float scale = ntgt / a;
  // rotated_i = z_i - 2*dew*w_i + 2*deu*q_i = kz*z_i + kq*zq_i
  float kz = 1.f - 2.f * dew / (wn * a);
  float kq = (2.f * deu - 2.f * dew / wn) / bb;
  float4 o;
  o.x = (kz * zv.x + kq * qv.x) * scale;
  o.y = (kz * zv.y + kq * qv.y) * scale;
  o.z = (kz * zv.z + kq * qv.z) * scale;
  o.w = (kz * zv.w + kq * qv.w) * scale;
  ((float4*)(zq_out + (size_t)row * Dd))[lane] = o;

  if (lane == 0) commit_ws[row] = szz - 2.f * szq + sqq;  // row-wise ||z - zq||^2
}

// ---------------------------------------------------------------------------
// Kernel 7: deterministic commit reduction (single block tree)
// ---------------------------------------------------------------------------
__global__ void k_commit(const float* __restrict__ commit_ws, float* __restrict__ out) {
  __shared__ float sm[256];
  int tid = threadIdx.x;
  float s = 0.f;
  for (int i = tid; i < NROWS; i += 256) s += commit_ws[i];
  sm[tid] = s;
  __syncthreads();
  for (int off = 128; off >= 1; off >>= 1) {
    if (tid < off) sm[tid] += sm[tid + off];
    __syncthreads();
  }
  if (tid == 0) out[0] = sm[0] * COMMIT_SCALE;
}

// ---------------------------------------------------------------------------
extern "C" void kernel_launch(void* const* d_in, const int* in_sizes, int n_in,
                              void* d_out, int out_size, void* d_ws, size_t ws_size,
                              hipStream_t stream) {
  (void)in_sizes; (void)n_in; (void)out_size; (void)ws_size;
  const float* z     = (const float*)d_in[0];
  const float* fc    = (const float*)d_in[1];
  const float* cv    = (const float*)d_in[2];
  const float* cg    = (const float*)d_in[3];
  const float* cbias = (const float*)d_in[4];

  float* out        = (float*)d_out;
  float* zq_out     = out;                           // NROWS*Dd
  float* fidx_out   = out + (size_t)NROWS * Dd;      // NROWS (indices as float)
  float* commit_out = fidx_out + NROWS;              // 1

  char* p = (char*)d_ws;
  float*  w    = (float*)p;  p += (size_t)Dd * Dd * 4;
  float*  cb   = (float*)p;  p += (size_t)Cc * Dd * 4;
  float*  cbn2 = (float*)p;  p += (size_t)(Cc + 16) * 4;            // +16 pad for prefetch
  bf16_t* cbh  = (bf16_t*)p; p += (size_t)(Cc + 16) * Dd * 2;       // +1 tile pad for prefetch
  bf16_t* zh   = (bf16_t*)p; p += (size_t)NROWS * Dd * 2;
  int*    idxw = (int*)p;    p += (size_t)NROWS * 4;
  float*  cws  = (float*)p;

  k_build_w<<<1, 128, 0, stream>>>(cv, cg, w);
  k_zconv<<<(NROWS * Dd) / 256, 256, 0, stream>>>(z, zh);
  k_build_cb<<<(Cc * Dd) / 256, 256, 0, stream>>>(fc, w, cbias, cb, cbh);
  k_cbn2<<<Cc / 8, 256, 0, stream>>>(cb, cbn2);
  k_argmin<<<NROWS / (16 * NT), 32, 0, stream>>>(zh, cbh, cbn2, idxw, fidx_out);
  k_finalize<<<NROWS / 8, 256, 0, stream>>>(z, cb, idxw, zq_out, cws);
  k_commit<<<1, 256, 0, stream>>>(cws, commit_out);
}
